// CausalSelfAttention_38285338476545
// MI455X (gfx1250) — matrix operand order repaired and checked by reading, not compiled, not run on hardware
//
#include <hip/hip_runtime.h>

typedef _Float16 v16h __attribute__((ext_vector_type(16)));
typedef _Float16 v8h  __attribute__((ext_vector_type(8)));
typedef float    v8f  __attribute__((ext_vector_type(8)));
typedef float    v4f  __attribute__((ext_vector_type(4)));
typedef unsigned int u32x4 __attribute__((ext_vector_type(4)));
typedef int          i32x8 __attribute__((ext_vector_type(8)));
typedef int          i32x4 __attribute__((ext_vector_type(4)));

#define B_   4
#define S_   2048
#define D_   1024
#define NROW (B_ * S_)   // 8192 rows of x

// LDS W-tile row stride: 32 f16 (64B = 16 DW) + TDM pad of 4 DW => 80 B = 40 f16
#define WT_ROW 40

static __device__ inline v8f wmma_f16(v16h a, v16h b, v8f c) {
  return __builtin_amdgcn_wmma_f32_16x16x32_f16(false, a, false, b,
                                                (short)0, c, false, false);
}

// ---------------------------------------------------------------------------
// TDM 2D tile load: global (row-major, rowStride elems) -> LDS, padded rows.
// D# layout per CDNA5 ISA 8.3/8.4. 6-arg clang-23 builtin form:
//   (u32x4 g0, i32x8 g1, i32x4 g2, i32x4 g3, i32x8 extra, i32 cpol)
// ---------------------------------------------------------------------------
#if __has_builtin(__builtin_amdgcn_tensor_load_to_lds)
#define HAVE_TDM 1
static __device__ inline void tdm_load_2d_f16(uint32_t lds_addr, const void* gaddr,
                                              uint32_t tile_x, uint32_t tile_y,
                                              uint32_t tensor_d0, uint32_t tensor_d1,
                                              uint64_t row_stride_elems) {
  const uint64_t ga = (uint64_t)(uintptr_t)gaddr;
  u32x4 g0;
  g0[0] = 1u;                                            // count=1, user mode
  g0[1] = lds_addr;                                      // LDS byte address
  g0[2] = (uint32_t)(ga & 0xFFFFFFFFu);                  // global_addr[31:0]
  g0[3] = (uint32_t)((ga >> 32) & 0x1FFFFFFu) | (2u << 30);  // addr[56:32], type=2
  i32x8 g1;
  // data_size=1 (2B), pad_enable=1, pad_interval=3 (16 DW), pad_amount=3 (4 DW)
  g1[0] = (int)((1u << 16) | (1u << 20) | (3u << 22) | (3u << 25));
  g1[1] = (int)((tensor_d0 & 0xFFFFu) << 16);            // dim0[15:0] @ [63:48]
  g1[2] = (int)(((tensor_d0 >> 16) & 0xFFFFu) | ((tensor_d1 & 0xFFFFu) << 16));
  g1[3] = (int)(((tensor_d1 >> 16) & 0xFFFFu) | ((tile_x & 0xFFFFu) << 16));
  g1[4] = (int)(tile_y & 0xFFFFu);                       // tile_dim1; tile_dim2=0
  g1[5] = (int)(uint32_t)(row_stride_elems & 0xFFFFFFFFu);   // dim0_stride[31:0]
  g1[6] = (int)(uint32_t)((row_stride_elems >> 32) & 0xFFFFu);
  g1[7] = 0;
  const i32x4 z4 = {0, 0, 0, 0};
  const i32x8 z8 = {0, 0, 0, 0, 0, 0, 0, 0};
  __builtin_amdgcn_tensor_load_to_lds(g0, g1, z4, z4, z8, 0);
}
static __device__ inline void tdm_wait(int n) {
  if (n) __builtin_amdgcn_s_wait_tensorcnt(1);
  else   __builtin_amdgcn_s_wait_tensorcnt(0);
}
#else
#define HAVE_TDM 0
#endif

// ---------------------------------------------------------------------------
// f32 -> f16 bulk convert (x and the three weight matrices), 8 elems/thread
// ---------------------------------------------------------------------------
__global__ __launch_bounds__(256) void cvt_kernel(const float* __restrict__ src,
                                                  _Float16* __restrict__ dst,
                                                  int n8) {
  const int i = blockIdx.x * 256 + threadIdx.x;
  if (i >= n8) return;
  v4f a = *(const v4f*)(src + (size_t)i * 8);
  v4f b = *(const v4f*)(src + (size_t)i * 8 + 4);
  v8h h;
  #pragma unroll
  for (int j = 0; j < 4; ++j) { h[j] = (_Float16)a[j]; h[4 + j] = (_Float16)b[j]; }
  *(v8h*)(dst + (size_t)i * 8) = h;
}

// ---------------------------------------------------------------------------
// QKV projection, f16 in / f16 out: out[m][n] = sum_k x[m][k]*W[n][k].
// grid (NROW/128, D/64, 3), block 256 = 8 waves; wave owns 16 rows x 64 cols.
// Shared 64x32 W tile staged in LDS by TDM (double-buffered, padded rows).
// ---------------------------------------------------------------------------
__global__ __launch_bounds__(256) void qkv_proj_kernel(
    const _Float16* __restrict__ xh,
    const _Float16* __restrict__ Wqh,
    const _Float16* __restrict__ Wkh,
    const _Float16* __restrict__ Wvh,
    _Float16* __restrict__ Qb,
    _Float16* __restrict__ Kb,
    _Float16* __restrict__ Vt)  // Vt: [b][d][s]
{
  __shared__ __align__(16) _Float16 Wt[2][64 * WT_ROW];   // 2 x 5 KB

  const int lane = threadIdx.x & 31;
  const int wave = threadIdx.x >> 5;
  const int hi   = (lane & 16) ? 1 : 0;
  const int l15  = lane & 15;

  const _Float16* W = (blockIdx.z == 0) ? Wqh : (blockIdx.z == 1) ? Wkh : Wvh;

  const int m0 = blockIdx.x * 128 + wave * 16;
  const int n0 = blockIdx.y * 64;
  const _Float16* xrow = xh + (size_t)(m0 + l15) * D_;

  v8f zero = {};
  v8f acc[4];
  #pragma unroll
  for (int t = 0; t < 4; ++t) acc[t] = zero;

  const int NSTEP = D_ / 32;

#if HAVE_TDM
  if (wave == 0 && lane == 0) {
    tdm_load_2d_f16((uint32_t)(uintptr_t)(void*)&Wt[0][0], W + (size_t)n0 * D_,
                    /*tile_x=*/32, /*tile_y=*/64, D_, 1u << 20, D_);
  }
#endif

  for (int step = 0; step < NSTEP; ++step) {
    const int kb = step * 32;
    const int buf = step & 1;

#if HAVE_TDM
    if (wave == 0 && lane == 0) {
      if (step + 1 < NSTEP) {
        tdm_load_2d_f16((uint32_t)(uintptr_t)(void*)&Wt[buf ^ 1][0],
                        W + (size_t)n0 * D_ + (kb + 32),
                        32, 64, D_, 1u << 20, D_);
        tdm_wait(1);   // current tile done, next may be in flight
      } else {
        tdm_wait(0);
      }
    }
#else
    // fallback: synchronous cooperative stage (8 rows per wave)
    {
      const int r = wave * 8 + (lane >> 2);          // 0..63 over 256 threads
      const int c = (lane & 3) * 8;                  // 8 f16 per thread
      v8h w8 = *(const v8h*)(W + (size_t)(n0 + r) * D_ + kb + c);
      *(v8h*)(&Wt[buf][r * WT_ROW + c]) = w8;
    }
#endif
    __syncthreads();   // W tile visible to all waves

    // ---- A fragment straight from global f16 ----
    const int a0 = kb + (hi ? 16 : 0);
    v8h alo = *(const v8h*)(xrow + a0);
    v8h aup = *(const v8h*)(xrow + a0 + 8);
    v16h afrag;
    #pragma unroll
    for (int i = 0; i < 8; ++i) { afrag[i] = alo[i]; afrag[8 + i] = aup[i]; }

    // ---- 4 B fragments from LDS (padded rows -> conflict-free b128) ----
    #pragma unroll
    for (int t = 0; t < 4; ++t) {
      const _Float16* bp = &Wt[buf][(t * 16 + l15) * WT_ROW + (hi ? 16 : 0)];
      v8h b0 = *(const v8h*)(bp);
      v8h b1 = *(const v8h*)(bp + 8);
      v16h bfrag;
      #pragma unroll
      for (int i = 0; i < 8; ++i) { bfrag[i] = b0[i]; bfrag[8 + i] = b1[i]; }
      acc[t] = wmma_f16(afrag, bfrag, acc[t]);
    }
    __syncthreads();   // all reads of this buffer done before it is re-filled
  }

  // ---- store f16 results (C layout: vgpr r -> M=r(+8 for hi), N=l15) ----
  #pragma unroll
  for (int t = 0; t < 4; ++t) {
    #pragma unroll
    for (int r = 0; r < 8; ++r) {
      const int row = m0 + r + (hi ? 8 : 0);
      const int col = n0 + t * 16 + l15;
      const _Float16 hv = (_Float16)acc[t][r];
      if (blockIdx.z == 0) {
        Qb[(size_t)row * D_ + col] = hv;
      } else if (blockIdx.z == 1) {
        Kb[(size_t)row * D_ + col] = hv;
      } else {
        const int b = row >> 11;
        const int s = row & (S_ - 1);
        Vt[((size_t)b * D_ + col) * S_ + s] = hv;
      }
    }
  }
}

// ---------------------------------------------------------------------------
// Flash attention: grid (S/16, B), block 256 = 8 waves; wave w owns D-slice
// [w*128, w*128+128). ds_add_f32 cross-wave score reduce, online softmax,
// P transposed via LDS, P*V WMMAs. Prefetch next K/V tiles each step.
// ---------------------------------------------------------------------------
__global__ __launch_bounds__(256) void attn_kernel(
    const _Float16* __restrict__ Qb,
    const _Float16* __restrict__ Kb,
    const _Float16* __restrict__ Vt,
    float* __restrict__ out)
{
  __shared__ float    S_lds[16 * 32];
  __shared__ __align__(16) _Float16 P_lds[16 * 32];

  const int lane = threadIdx.x & 31;
  const int wave = threadIdx.x >> 5;
  const int hi   = (lane & 16) ? 1 : 0;
  const int l15  = lane & 15;
  const int rowoff = hi ? 8 : 0;

  const int q0 = blockIdx.x * 16;
  const int b  = blockIdx.y;
  const int d0 = wave * 128;

  v16h qfrag[4];
  {
    const _Float16* qr = Qb + (size_t)(b * S_ + q0 + l15) * D_;
    #pragma unroll
    for (int kk = 0; kk < 4; ++kk) {
      const int a0 = d0 + kk * 32 + (hi ? 16 : 0);
      v8h lo = *(const v8h*)(qr + a0);
      v8h up = *(const v8h*)(qr + a0 + 8);
      v16h f;
      #pragma unroll
      for (int i = 0; i < 8; ++i) { f[i] = lo[i]; f[8 + i] = up[i]; }
      qfrag[kk] = f;
    }
  }

  v8f zero = {};
  v8f o[8];
  #pragma unroll
  for (int t = 0; t < 8; ++t) o[t] = zero;
  float mrow[8], lrow[8];
  #pragma unroll
  for (int r = 0; r < 8; ++r) { mrow[r] = -3.0e38f; lrow[r] = 0.f; }

  const float inv_sqrt_d = 0.03125f;  // 1/sqrt(1024)
  const int kend = q0 + 16;           // causal bound (uniform per block)

  for (int kb = 0; kb < kend; kb += 32) {
    // prefetch next key step's K rows and V columns into cache
    if (kb + 32 < kend) {
      __builtin_prefetch(Kb + (size_t)(b * S_ + kb + 32 + l15) * D_ + d0, 0, 1);
      __builtin_prefetch(Vt + ((size_t)b * D_ + d0 + l15) * S_ + kb + 32, 0, 1);
    }

    // ---- partial 16x32 score tile over this wave's 128-wide D slice ----
    v8f s0 = zero, s1 = zero;
    #pragma unroll
    for (int kk = 0; kk < 4; ++kk) {
      const int dro = d0 + kk * 32 + (hi ? 16 : 0);
      const _Float16* kr0 = Kb + (size_t)(b * S_ + kb + l15) * D_ + dro;
      const _Float16* kr1 = Kb + (size_t)(b * S_ + kb + 16 + l15) * D_ + dro;
      v8h x0 = *(const v8h*)(kr0), x1 = *(const v8h*)(kr0 + 8);
      v8h y0 = *(const v8h*)(kr1), y1 = *(const v8h*)(kr1 + 8);
      v16h kf0, kf1;
      #pragma unroll
      for (int i = 0; i < 8; ++i) {
        kf0[i] = x0[i]; kf0[8 + i] = x1[i];
        kf1[i] = y0[i]; kf1[8 + i] = y1[i];
      }
      s0 = wmma_f16(qfrag[kk], kf0, s0);
      s1 = wmma_f16(qfrag[kk], kf1, s1);
    }

    // ---- cross-wave reduce into LDS (wave0 stores, others ds_add_f32) ----
    if (wave == 0) {
      #pragma unroll
      for (int r = 0; r < 8; ++r) {
        S_lds[(r + rowoff) * 32 + l15]      = s0[r];
        S_lds[(r + rowoff) * 32 + 16 + l15] = s1[r];
      }
    }
    __syncthreads();
    if (wave != 0) {
      #pragma unroll
      for (int r = 0; r < 8; ++r) {
        atomicAdd(&S_lds[(r + rowoff) * 32 + l15],      s0[r]);
        atomicAdd(&S_lds[(r + rowoff) * 32 + 16 + l15], s1[r]);
      }
    }
    __syncthreads();

    // ---- online softmax (redundant per wave; rows live in lane halves) ----
    float p0[8], p1[8], corr[8];
    {
      float t0[8], t1[8], mx[8];
      #pragma unroll
      for (int r = 0; r < 8; ++r) {
        const int row = r + rowoff;
        float a = S_lds[row * 32 + l15]      * inv_sqrt_d;
        float c = S_lds[row * 32 + 16 + l15] * inv_sqrt_d;
        if (kb + l15      > q0 + row) a = -3.0e38f;   // causal mask
        if (kb + 16 + l15 > q0 + row) c = -3.0e38f;
        t0[r] = a; t1[r] = c;
        mx[r] = fmaxf(a, c);
      }
      #pragma unroll
      for (int r = 0; r < 8; ++r)
        #pragma unroll
        for (int msk = 8; msk >= 1; msk >>= 1)
          mx[r] = fmaxf(mx[r], __shfl_xor(mx[r], msk, 32));
      #pragma unroll
      for (int r = 0; r < 8; ++r) {
        const float mnew = fmaxf(mrow[r], mx[r]);
        p0[r]   = __expf(t0[r] - mnew);
        p1[r]   = __expf(t1[r] - mnew);
        corr[r] = __expf(mrow[r] - mnew);
        float sum = p0[r] + p1[r];
        #pragma unroll
        for (int msk = 8; msk >= 1; msk >>= 1)
          sum += __shfl_xor(sum, msk, 32);
        lrow[r] = lrow[r] * corr[r] + sum;
        mrow[r] = mnew;
      }
    }

    // ---- P tile to LDS (identical across waves; wave0 writes) ----
    if (wave == 0) {
      #pragma unroll
      for (int r = 0; r < 8; ++r) {
        P_lds[(r + rowoff) * 32 + l15]      = (_Float16)p0[r];
        P_lds[(r + rowoff) * 32 + 16 + l15] = (_Float16)p1[r];
      }
    }
    __syncthreads();

    // ---- read P back as A-fragment (C-layout -> A-layout via LDS) ----
    v16h pfrag;
    {
      const _Float16* pr = &P_lds[l15 * 32 + (hi ? 16 : 0)];
      v8h lo = *(const v8h*)(pr);
      v8h up = *(const v8h*)(pr + 8);
      #pragma unroll
      for (int i = 0; i < 8; ++i) { pfrag[i] = lo[i]; pfrag[8 + i] = up[i]; }
    }

    // ---- rescale O then accumulate P*V over this wave's D slice ----
    #pragma unroll
    for (int t = 0; t < 8; ++t)
      #pragma unroll
      for (int r = 0; r < 8; ++r)
        o[t][r] *= corr[r];

    #pragma unroll
    for (int t = 0; t < 8; ++t) {
      const int dcol = d0 + t * 16 + l15;
      const _Float16* vr = Vt + ((size_t)b * D_ + dcol) * S_ + kb + (hi ? 16 : 0);
      v8h x0 = *(const v8h*)(vr), x1 = *(const v8h*)(vr + 8);
      v16h vfrag;
      #pragma unroll
      for (int i = 0; i < 8; ++i) { vfrag[i] = x0[i]; vfrag[8 + i] = x1[i]; }
      o[t] = wmma_f16(pfrag, vfrag, o[t]);
    }
    __syncthreads();   // protect S_lds/P_lds reuse next iteration
  }

  // ---- normalize and store f32 output ----
  #pragma unroll
  for (int t = 0; t < 8; ++t) {
    const int dcol = d0 + t * 16 + l15;
    #pragma unroll
    for (int r = 0; r < 8; ++r) {
      const int row = q0 + r + rowoff;
      out[(size_t)(b * S_ + row) * D_ + dcol] = o[t][r] / lrow[r];
    }
  }
}

// ---------------------------------------------------------------------------
extern "C" void kernel_launch(void* const* d_in, const int* in_sizes, int n_in,
                              void* d_out, int out_size, void* d_ws, size_t ws_size,
                              hipStream_t stream) {
  const float* x  = (const float*)d_in[0];
  const float* Wq = (const float*)d_in[1];
  const float* Wk = (const float*)d_in[2];
  const float* Wv = (const float*)d_in[3];
  float* out = (float*)d_out;

  // workspace layout (f16):
  //   Qb [NROW,D] | Kb [NROW,D] | Vt [B,D,S] | xh [NROW,D] | Wqh | Wkh | Wvh
  _Float16* Qb  = (_Float16*)d_ws;
  _Float16* Kb  = Qb  + (size_t)NROW * D_;
  _Float16* Vt  = Kb  + (size_t)NROW * D_;
  _Float16* xh  = Vt  + (size_t)NROW * D_;
  _Float16* Wqh = xh  + (size_t)NROW * D_;
  _Float16* Wkh = Wqh + (size_t)D_ * D_;
  _Float16* Wvh = Wkh + (size_t)D_ * D_;

  const int n8x = NROW * D_ / 8;     // 1048576
  const int n8w = D_ * D_ / 8;       // 131072
  cvt_kernel<<<dim3((n8x + 255) / 256), 256, 0, stream>>>(x,  xh,  n8x);
  cvt_kernel<<<dim3((n8w + 255) / 256), 256, 0, stream>>>(Wq, Wqh, n8w);
  cvt_kernel<<<dim3((n8w + 255) / 256), 256, 0, stream>>>(Wk, Wkh, n8w);
  cvt_kernel<<<dim3((n8w + 255) / 256), 256, 0, stream>>>(Wv, Wvh, n8w);

  qkv_proj_kernel<<<dim3(NROW / 128, D_ / 64, 3), 256, 0, stream>>>(
      xh, Wqh, Wkh, Wvh, Qb, Kb, Vt);
  attn_kernel<<<dim3(S_ / 16, B_), 256, 0, stream>>>(Qb, Kb, Vt, out);
}
